// LoRAQuantLinear_16234976379141
// MI455X (gfx1250) — matrix operand order repaired and checked by reading, not compiled, not run on hardware
//
#include <hip/hip_runtime.h>
#include <hip/hip_bf16.h>
#include <stdint.h>

typedef __bf16 bf16_t;
typedef __attribute__((ext_vector_type(16))) __bf16 v16bf;
typedef __attribute__((ext_vector_type(8)))  float  v8f;
typedef __attribute__((ext_vector_type(4))) unsigned int v4u;
typedef __attribute__((ext_vector_type(8))) int          v8i;
typedef __attribute__((ext_vector_type(4))) int          v4i;

#define BM 128
#define BN 128
#define BK 64              // two 16x16x32 WMMA k-steps per staged tile
#define PADF 4             // TDM pad: 4 DWORDs after every 64 -> row stride 68 f32
#define ROWF (BK + PADF)
#define TPB 256
#define RMAX 16

#if __has_builtin(__builtin_amdgcn_tensor_load_to_lds)
#define HAVE_TDM 1
#else
#define HAVE_TDM 0
#endif

// ---------------------------------------------------------------------------
// TDM: issue one 2D tile load (rows x BK f32) global -> LDS, padded rows.
// D# packed per CDNA5 ISA 8.3/8.4 (group0 128b, group1 256b, groups 2/3 = 0).
// This toolchain exposes the 6-arg builtin (g0, g1, g2, g3, extra, cpol).
// ---------------------------------------------------------------------------
__device__ __forceinline__ void tdm_load_tile_f32(const float* gsrc,
                                                  uint32_t lds_byte_off,
                                                  int Kdim, int rows) {
#if HAVE_TDM
  const uint64_t ga = (uint64_t)(uintptr_t)gsrc;
  v4u g0;
  g0[0] = 1u;                                        // count=1, user mode
  g0[1] = lds_byte_off;                              // lds_addr
  g0[2] = (uint32_t)ga;                              // global_addr[31:0]
  g0[3] = (uint32_t)((ga >> 32) & 0x1FFFFFFu)        // global_addr[56:32]
        | (2u << 30);                                // type = 2 ("image")
  const uint32_t td0 = (uint32_t)Kdim;               // tensor_dim0 (elems)
  const uint32_t td1 = 1u << 30;                     // tensor_dim1 (no OOB)
  v8i g1;
  g1[0] = (int)((2u << 16)                           // data_size = 4B
              | (1u << 20)                           // pad_enable
              | (5u << 22)                           // pad_interval: 64 DWORDs
              | (3u << 25));                         // pad_amount: 4 DWORDs
  g1[1] = (int)((td0 & 0xFFFFu) << 16);              // tensor_dim0[15:0]
  g1[2] = (int)(((td0 >> 16) & 0xFFFFu)              // tensor_dim0[31:16]
              | ((td1 & 0xFFFFu) << 16));            // tensor_dim1[15:0]
  g1[3] = (int)(((td1 >> 16) & 0xFFFFu)              // tensor_dim1[31:16]
              | ((uint32_t)BK << 16));               // tile_dim0 = 64
  g1[4] = (int)(uint32_t)rows;                       // tile_dim1 (tile_dim2=0)
  g1[5] = (int)(uint32_t)Kdim;                       // tensor_dim0_stride[31:0]
  g1[6] = 0;                                         // stride0[47:32], stride1 lo
  g1[7] = 0;
  const v4i z4 = {0, 0, 0, 0};
  const v8i z8 = {0, 0, 0, 0, 0, 0, 0, 0};
  __builtin_amdgcn_tensor_load_to_lds(g0, g1, z4, z4, z8, 0);
#endif
}

// ---------------------------------------------------------------------------
// Kernel 1: xa[m][r] = 2.0 * scale_a * sum_k x[m][k] * qa[r][k]
// ---------------------------------------------------------------------------
#define K1_ROWS 16
#define K1_BK   256

__global__ __launch_bounds__(TPB)
void lora_xa_kernel(const float* __restrict__ x,
                    const int*   __restrict__ qa,
                    const float* __restrict__ scale_a_p,
                    float*       __restrict__ xa,
                    int M, int Kdim, int R) {
  __shared__ float xs[K1_ROWS][K1_BK + 1];
  __shared__ float as[RMAX][K1_BK + 1];

  const int tid = threadIdx.x;
  const int bm0 = blockIdx.x * K1_ROWS;
  const float sa = scale_a_p[0] * 2.0f;   // fold LORA_SCALE

  const int m = tid >> 4;
  const int r = tid & 15;
  float acc = 0.0f;

  for (int k0 = 0; k0 < Kdim; k0 += K1_BK) {
    float4 xv[4];
    int4   qv[4];
    // phase 1: issue all global loads (MLP)
#pragma unroll
    for (int i = 0; i < 4; ++i) {
      int idx = tid + i * TPB;
      int row = idx >> 6;
      int c   = (idx & 63) << 2;
      xv[i] = *(const float4*)(x + (size_t)(bm0 + row) * Kdim + k0 + c);
      qv[i] = (row < R) ? *(const int4*)(qa + (size_t)row * Kdim + k0 + c)
                        : make_int4(0, 0, 0, 0);
    }
    // phase 2: stores to LDS
#pragma unroll
    for (int i = 0; i < 4; ++i) {
      int idx = tid + i * TPB;
      int row = idx >> 6;
      int c   = (idx & 63) << 2;
      xs[row][c + 0] = xv[i].x; xs[row][c + 1] = xv[i].y;
      xs[row][c + 2] = xv[i].z; xs[row][c + 3] = xv[i].w;
      if (row < R) {
        as[row][c + 0] = (float)qv[i].x; as[row][c + 1] = (float)qv[i].y;
        as[row][c + 2] = (float)qv[i].z; as[row][c + 3] = (float)qv[i].w;
      }
    }
    __syncthreads();
    if (r < R) {
#pragma unroll 8
      for (int k = 0; k < K1_BK; ++k)
        acc = fmaf(xs[m][k], as[r][k], acc);
    }
    __syncthreads();
  }
  if (r < R)
    xa[(size_t)(bm0 + m) * R + r] = acc * sa;
}

// ---------------------------------------------------------------------------
// Kernel 2: out = x @ W^T + bias + xa @ (scale_b*qb)^T
// TDM async tile staging (f32, padded rows), bf16 WMMA with f32 accum,
// fused rank-R LoRA + bias epilogue.
// ---------------------------------------------------------------------------
__global__ __launch_bounds__(TPB)
void lora_gemm_kernel(const float* __restrict__ x,
                      const float* __restrict__ W,
                      const float* __restrict__ bias,
                      const int*   __restrict__ qb,
                      const float* __restrict__ scale_b_p,
                      const float* __restrict__ xa,
                      float*       __restrict__ out,
                      int M, int N, int Kdim, int R) {
  __shared__ __align__(16) float Asf[2][BM][ROWF];   // 2 x 128 x 68 f32
  __shared__ __align__(16) float Bsf[2][BN][ROWF];
  __shared__ float xaS[BM][RMAX];
  __shared__ float bS [BN][RMAX];

  const int tid   = threadIdx.x;
  const int lane  = tid & 31;
  const int wid   = tid >> 5;          // 8 waves
  const int waveM = wid >> 2;          // 0..1 -> 64 rows
  const int waveN = wid & 3;           // 0..3 -> 32 cols
  const int bm0   = blockIdx.y * BM;
  const int bn0   = blockIdx.x * BN;
  const float sb  = scale_b_p[0];

  // ---- per-wave TDM issue: each wave stages its own 16-row slice of A & B
  auto issueTiles = [&](int buf, int kb) {
#if HAVE_TDM
    const float* ga = x + (size_t)(bm0 + wid * 16) * Kdim + kb;
    tdm_load_tile_f32(ga, (uint32_t)(uintptr_t)&Asf[buf][wid * 16][0], Kdim, 16);
    const float* gb = W + (size_t)(bn0 + wid * 16) * Kdim + kb;
    tdm_load_tile_f32(gb, (uint32_t)(uintptr_t)&Bsf[buf][wid * 16][0], Kdim, 16);
#else
    // fallback: cooperative copy of this wave's slices (lane-parallel)
#pragma unroll
    for (int i = 0; i < 8; ++i) {
      int idx = lane + i * 32;               // float4 idx within 16xBK slice
      int row = idx >> 4;                    // 16 float4 per row (BK=64)
      int c   = (idx & 15) << 2;
      float4 va = *(const float4*)(x + (size_t)(bm0 + wid * 16 + row) * Kdim + kb + c);
      float4 vb = *(const float4*)(W + (size_t)(bn0 + wid * 16 + row) * Kdim + kb + c);
      *(float4*)(&Asf[buf][wid * 16 + row][c]) = va;
      *(float4*)(&Bsf[buf][wid * 16 + row][c]) = vb;
    }
#endif
  };

  // ---- fragment builders: f32 LDS -> bf16 fragment (cvt co-executes w/ WMMA)
  auto loadA = [&](int buf, int mt, int ks) -> v16bf {
    int ml = waveM * 64 + mt * 16 + (lane & 15);
    int k0 = ks * 32 + ((lane < 16) ? 0 : 8);
    const float* p = &Asf[buf][ml][0];
    float4 a0 = *(const float4*)(p + k0);
    float4 a1 = *(const float4*)(p + k0 + 4);
    float4 a2 = *(const float4*)(p + k0 + 16);
    float4 a3 = *(const float4*)(p + k0 + 20);
    v16bf r;
    r[0]  = (bf16_t)a0.x; r[1]  = (bf16_t)a0.y; r[2]  = (bf16_t)a0.z; r[3]  = (bf16_t)a0.w;
    r[4]  = (bf16_t)a1.x; r[5]  = (bf16_t)a1.y; r[6]  = (bf16_t)a1.z; r[7]  = (bf16_t)a1.w;
    r[8]  = (bf16_t)a2.x; r[9]  = (bf16_t)a2.y; r[10] = (bf16_t)a2.z; r[11] = (bf16_t)a2.w;
    r[12] = (bf16_t)a3.x; r[13] = (bf16_t)a3.y; r[14] = (bf16_t)a3.z; r[15] = (bf16_t)a3.w;
    return r;
  };
  auto loadB = [&](int buf, int nt, int ks) -> v16bf {
    int nl = waveN * 32 + nt * 16 + (lane & 15);
    int k0 = ks * 32 + ((lane < 16) ? 0 : 8);
    const float* p = &Bsf[buf][nl][0];
    float4 b0 = *(const float4*)(p + k0);
    float4 b1 = *(const float4*)(p + k0 + 4);
    float4 b2 = *(const float4*)(p + k0 + 16);
    float4 b3 = *(const float4*)(p + k0 + 20);
    v16bf r;
    r[0]  = (bf16_t)b0.x; r[1]  = (bf16_t)b0.y; r[2]  = (bf16_t)b0.z; r[3]  = (bf16_t)b0.w;
    r[4]  = (bf16_t)b1.x; r[5]  = (bf16_t)b1.y; r[6]  = (bf16_t)b1.z; r[7]  = (bf16_t)b1.w;
    r[8]  = (bf16_t)b2.x; r[9]  = (bf16_t)b2.y; r[10] = (bf16_t)b2.z; r[11] = (bf16_t)b2.w;
    r[12] = (bf16_t)b3.x; r[13] = (bf16_t)b3.y; r[14] = (bf16_t)b3.z; r[15] = (bf16_t)b3.w;
    return r;
  };

  v8f acc[4][2];
#pragma unroll
  for (int mt = 0; mt < 4; ++mt)
#pragma unroll
    for (int nt = 0; nt < 2; ++nt)
#pragma unroll
      for (int i = 0; i < 8; ++i) acc[mt][nt][i] = 0.0f;

  // ---- prologue: kick off first tiles, stage LoRA tiles under the DMA
  issueTiles(0, 0);
  for (int idx = tid; idx < BM * R; idx += TPB) {
    int row = idx / R, c = idx % R;
    xaS[row][c] = xa[(size_t)(bm0 + row) * R + c];
    bS [row][c] = (float)qb[(size_t)(bn0 + row) * R + c] * sb;
  }
#if HAVE_TDM
  __builtin_amdgcn_s_wait_tensorcnt(0);
#endif
  __syncthreads();

  // ---- main loop: DMA next tile while computing current
  const int NT = Kdim / BK;
  for (int kt = 0; kt < NT; ++kt) {
    const int cur = kt & 1;
    if (kt + 1 < NT) issueTiles(cur ^ 1, (kt + 1) * BK);

#pragma unroll
    for (int ks = 0; ks < 2; ++ks) {
      v16bf af[4], bfr[2];
#pragma unroll
      for (int mt = 0; mt < 4; ++mt) af[mt] = loadA(cur, mt, ks);
#pragma unroll
      for (int nt = 0; nt < 2; ++nt) bfr[nt] = loadB(cur, nt, ks);
#pragma unroll
      for (int mt = 0; mt < 4; ++mt)
#pragma unroll
        for (int nt = 0; nt < 2; ++nt)
          acc[mt][nt] = __builtin_amdgcn_wmma_f32_16x16x32_bf16(
              false, af[mt], false, bfr[nt], (short)0, acc[mt][nt],
              false, false);
    }
#if HAVE_TDM
    __builtin_amdgcn_s_wait_tensorcnt(0);
#endif
    __syncthreads();
  }

  // ---- epilogue: bias + rank-R LoRA correction, store
  const int nQ = lane & 15;
  const int mH = (lane >> 4) << 3;

  float bv[2][RMAX];
  float biasv[2];
#pragma unroll
  for (int nt = 0; nt < 2; ++nt) {
    int nl = waveN * 32 + nt * 16 + nQ;
    biasv[nt] = bias[bn0 + nl];
    for (int r = 0; r < R; ++r) bv[nt][r] = bS[nl][r];
  }

#pragma unroll
  for (int mt = 0; mt < 4; ++mt) {
#pragma unroll
    for (int v = 0; v < 8; ++v) {
      int ml = waveM * 64 + mt * 16 + mH + v;
      float l0 = 0.0f, l1 = 0.0f;
      for (int r = 0; r < R; ++r) {
        float xv = xaS[ml][r];
        l0 = fmaf(xv, bv[0][r], l0);
        l1 = fmaf(xv, bv[1][r], l1);
      }
      size_t base = (size_t)(bm0 + ml) * N + bn0 + waveN * 32 + nQ;
      out[base]      = acc[mt][0][v] + biasv[0] + l0;
      out[base + 16] = acc[mt][1][v] + biasv[1] + l1;
    }
  }
}

// ---------------------------------------------------------------------------
extern "C" void kernel_launch(void* const* d_in, const int* in_sizes, int n_in,
                              void* d_out, int out_size, void* d_ws, size_t ws_size,
                              hipStream_t stream) {
  const float* x    = (const float*)d_in[0];
  const float* W    = (const float*)d_in[1];
  const float* bias = (const float*)d_in[2];
  const int*   qa   = (const int*)d_in[3];
  const int*   qb   = (const int*)d_in[4];
  const float* sa   = (const float*)d_in[5];
  const float* sb   = (const float*)d_in[6];
  float*       out  = (float*)d_out;

  const int D_OUT = in_sizes[2];
  const int R     = in_sizes[4] / D_OUT;
  const int D_IN  = in_sizes[3] / R;
  const int M     = in_sizes[0] / D_IN;   // B*S

  float* xa = (float*)d_ws;               // M*R floats of scratch

  dim3 g1(M / K1_ROWS);
  lora_xa_kernel<<<g1, TPB, 0, stream>>>(x, qa, sa, xa, M, D_IN, R);

  dim3 g2(D_OUT / BN, M / BM);
  lora_gemm_kernel<<<g2, TPB, 0, stream>>>(x, W, bias, qb, sb, xa, out,
                                           M, D_OUT, D_IN, R);
}
